// VQEmbeddingEMA_45578192945826
// MI455X (gfx1250) — compile-verified, matrix operand order checked by reading, not run on hardware
//
#include <hip/hip_runtime.h>
#include <hip/hip_bf16.h>

typedef __attribute__((ext_vector_type(16))) _Float16 v16h;
typedef __attribute__((ext_vector_type(8)))  _Float16 v8h;
typedef __attribute__((ext_vector_type(4)))  _Float16 v4h;
typedef __attribute__((ext_vector_type(8)))  float    v8f;

#define EMB_DIM   256
#define N_EMB     1024
#define EGROUP    64           // embeddings staged in LDS per iteration
#define NGROUP    (N_EMB / EGROUP)   // 16
#define ROWSTRIDE 264          // padded f16 row stride (528B) to avoid LDS bank conflicts
#define COMMIT    0.25f

// ---------------------------------------------------------------------------
// Zero the accumulator region of workspace: counts[1024] + num + den
// ---------------------------------------------------------------------------
__global__ void vq_zero_kernel(float* acc) {
    for (int i = threadIdx.x; i < N_EMB + 2; i += blockDim.x) acc[i] = 0.0f;
}

// ---------------------------------------------------------------------------
// ||e_m||^2 per embedding row.  One wave per row.
// ---------------------------------------------------------------------------
__global__ __launch_bounds__(32) void vq_enorm_kernel(const float* __restrict__ emb,
                                                      float* __restrict__ enorm) {
    const int m = blockIdx.x;
    const int lane = threadIdx.x;
    float s = 0.0f;
#pragma unroll
    for (int j = 0; j < EMB_DIM / 32; ++j) {
        float v = emb[(size_t)m * EMB_DIM + lane + j * 32];
        s += v * v;
    }
#pragma unroll
    for (int off = 16; off >= 1; off >>= 1) s += __shfl_xor(s, off, 32);
    if (lane == 0) enorm[m] = s;
}

// ---------------------------------------------------------------------------
// Main kernel: argmin_m (||e_m||^2 - 2 x.e_m) via v_wmma_f32_16x16x32_f16.
// Block = 256 threads = 8 waves.  Each wave owns TWO 16-row x-tiles (A kept
// resident in 128 VGPRs for full K=256), so every LDS B-fragment feeds two
// independent WMMA accumulation chains (2x LDS-bandwidth reuse + better
// pipeline overlap).  Block covers 256 rows; grid = B/256.
// ---------------------------------------------------------------------------
__global__ __launch_bounds__(256) void vq_argmin_wmma_kernel(
    const float* __restrict__ x, const float* __restrict__ emb,
    const float* __restrict__ enorm, int* __restrict__ idx_out) {

    __shared__ _Float16 eLds[EGROUP * ROWSTRIDE];   // ~33 KB
    __shared__ float    nLds[EGROUP];

    const int tid  = threadIdx.x;
    const int lane = tid & 31;
    const int wave = tid >> 5;
    const int l16  = lane & 15;
    const int kh   = lane >> 4;          // which K half-block this lane holds
    const int rb   = blockIdx.x * 256 + wave * 32;   // wave's 32-row region

    // ---- Build A fragments for two 16x256 x-tiles (f32 -> f16) -----------
    // 16-bit A layout: lane l16 = M row; elements [0..7] = K base..base+7,
    // [8..15] = K base+16..base+23, base = kh*8 within each 32-wide K tile.
    v16h afrag[2][8];
#pragma unroll
    for (int t = 0; t < 2; ++t) {
        const float4* xrow = (const float4*)(x + (size_t)(rb + t * 16 + l16) * EMB_DIM);
#pragma unroll
        for (int kt = 0; kt < 8; ++kt) {
            const int k0 = kt * 32 + kh * 8;
            float4 f0 = xrow[k0 / 4];
            float4 f1 = xrow[k0 / 4 + 1];
            float4 f2 = xrow[(k0 + 16) / 4];
            float4 f3 = xrow[(k0 + 16) / 4 + 1];
            v16h a;
            a[0]  = (_Float16)f0.x; a[1]  = (_Float16)f0.y; a[2]  = (_Float16)f0.z; a[3]  = (_Float16)f0.w;
            a[4]  = (_Float16)f1.x; a[5]  = (_Float16)f1.y; a[6]  = (_Float16)f1.z; a[7]  = (_Float16)f1.w;
            a[8]  = (_Float16)f2.x; a[9]  = (_Float16)f2.y; a[10] = (_Float16)f2.z; a[11] = (_Float16)f2.w;
            a[12] = (_Float16)f3.x; a[13] = (_Float16)f3.y; a[14] = (_Float16)f3.z; a[15] = (_Float16)f3.w;
            afrag[t][kt] = a;
        }
    }

    float minv[2][8];
    int   mini[2][8];
#pragma unroll
    for (int t = 0; t < 2; ++t)
#pragma unroll
        for (int v = 0; v < 8; ++v) { minv[t][v] = 3.4e38f; mini[t][v] = 0; }

    // ---- Loop over embedding groups staged in LDS -------------------------
    const int r    = tid >> 2;           // staging row 0..63
    const int part = tid & 3;            // 64-float chunk within row

    for (int g = 0; g < NGROUP; ++g) {
        // cooperative stage: 64 embeddings f32 -> f16 into LDS
        const float4* src = (const float4*)(emb + (size_t)(g * EGROUP + r) * EMB_DIM + part * 64);
        v4h* dst = (v4h*)(&eLds[r * ROWSTRIDE + part * 64]);
#pragma unroll
        for (int j = 0; j < 16; ++j) {
            float4 f = src[j];
            v4h h; h[0] = (_Float16)f.x; h[1] = (_Float16)f.y; h[2] = (_Float16)f.z; h[3] = (_Float16)f.w;
            dst[j] = h;
        }
        if (tid < EGROUP) nLds[tid] = enorm[g * EGROUP + tid];
        if (g + 1 < NGROUP)
            __builtin_prefetch(emb + (size_t)((g + 1) * EGROUP + r) * EMB_DIM + part * 64, 0, 1);
        __syncthreads();

        // 4 column tiles of 16 embeddings each; each B fragment feeds 2 WMMAs
#pragma unroll
        for (int ct = 0; ct < 4; ++ct) {
            const int mloc = ct * 16 + l16;            // this lane's B column
            const v8h* erow = (const v8h*)(&eLds[mloc * ROWSTRIDE]);
            v8f c0 = {};
            v8f c1 = {};
#pragma unroll
            for (int kt = 0; kt < 8; ++kt) {
                v8h b0 = erow[kt * 4 + kh];            // K base..base+7
                v8h b1 = erow[kt * 4 + kh + 2];        // K base+16..base+23
                v16h b;
#pragma unroll
                for (int i = 0; i < 8; ++i) { b[i] = b0[i]; b[8 + i] = b1[i]; }
                c0 = __builtin_amdgcn_wmma_f32_16x16x32_f16(
                        false, afrag[0][kt], false, b, (short)0, c0, false, false);
                c1 = __builtin_amdgcn_wmma_f32_16x16x32_f16(
                        false, afrag[1][kt], false, b, (short)0, c1, false, false);
            }
            const int nglob = g * EGROUP + mloc;
            const float en = nLds[mloc];
#pragma unroll
            for (int v = 0; v < 8; ++v) {
                float s0 = en - 2.0f * c0[v];
                if (s0 < minv[0][v]) { minv[0][v] = s0; mini[0][v] = nglob; }
                float s1 = en - 2.0f * c1[v];
                if (s1 < minv[1][v]) { minv[1][v] = s1; mini[1][v] = nglob; }
            }
        }
        __syncthreads();
    }

    // ---- Cross-lane min-reduction over the 16 lanes sharing each M row ----
#pragma unroll
    for (int off = 1; off < 16; off <<= 1) {
#pragma unroll
        for (int t = 0; t < 2; ++t)
#pragma unroll
            for (int v = 0; v < 8; ++v) {
                float ov = __shfl_xor(minv[t][v], off, 32);
                int   oi = __shfl_xor(mini[t][v], off, 32);
                if (ov < minv[t][v] || (ov == minv[t][v] && oi < mini[t][v])) {
                    minv[t][v] = ov; mini[t][v] = oi;
                }
            }
    }
    if (l16 == 0) {
#pragma unroll
        for (int t = 0; t < 2; ++t)
#pragma unroll
            for (int v = 0; v < 8; ++v)
                idx_out[rb + t * 16 + kh * 8 + v] = mini[t][v];  // C layout: M = v + (lane/16)*8
    }
}

// ---------------------------------------------------------------------------
// Gather quantized rows, write output, masked commitment-loss partials, counts.
// One wave per row, 8 rows per block.
// ---------------------------------------------------------------------------
__global__ __launch_bounds__(256) void vq_gather_loss_kernel(
    const float* __restrict__ x, const float* __restrict__ emb,
    const int* __restrict__ idx, float* __restrict__ q_out,
    float* __restrict__ idx_out_f, float* __restrict__ counts,
    float* __restrict__ num, float* __restrict__ den) {

    __shared__ float bnum, bden;
    const int tid  = threadIdx.x;
    const int lane = tid & 31;
    const int wave = tid >> 5;
    const int row  = blockIdx.x * 8 + wave;

    if (tid == 0) { bnum = 0.0f; bden = 0.0f; }
    __syncthreads();

    const int id = idx[row];
    const float4* e4 = (const float4*)(emb + (size_t)id * EMB_DIM);
    const float4* x4 = (const float4*)(x + (size_t)row * EMB_DIM);
    float4* o4 = (float4*)(q_out + (size_t)row * EMB_DIM);

    float ss = 0.0f, sa = 0.0f;
#pragma unroll
    for (int j = 0; j < 2; ++j) {
        const int c = lane * 2 + j;
        float4 q = e4[c];
        float4 xv = x4[c];
        o4[c] = q;                                     // quantized_st forward == quantized
        float dx = xv.x - q.x, dy = xv.y - q.y, dz = xv.z - q.z, dw = xv.w - q.w;
        ss += dx * dx + dy * dy + dz * dz + dw * dw;
        sa += fabsf(xv.x) + fabsf(xv.y) + fabsf(xv.z) + fabsf(xv.w);
    }
#pragma unroll
    for (int off = 16; off >= 1; off >>= 1) {
        ss += __shfl_xor(ss, off, 32);
        sa += __shfl_xor(sa, off, 32);
    }
    if (lane == 0) {
        float np = (sa > 0.0f) ? 1.0f : 0.0f;
        atomicAdd(&counts[id], 1.0f);
        idx_out_f[row] = (float)id;
        atomicAdd(&bnum, np * ss * (1.0f / EMB_DIM));
        atomicAdd(&bden, np);
    }
    __syncthreads();
    if (tid == 0) {
        atomicAdd(num, bnum);
        atomicAdd(den, bden);
    }
}

// ---------------------------------------------------------------------------
// loss + perplexity scalars
// ---------------------------------------------------------------------------
__global__ __launch_bounds__(256) void vq_finalize_kernel(
    const float* __restrict__ counts, const float* __restrict__ num,
    const float* __restrict__ den, float* __restrict__ out_loss,
    float* __restrict__ out_perp, int Brows) {

    __shared__ float red[256];
    const int t = threadIdx.x;
    float h = 0.0f;
    const float invB = 1.0f / (float)Brows;
    for (int i = t; i < N_EMB; i += 256) {
        float p = counts[i] * invB;
        h -= p * logf(p + 1e-10f);
    }
    red[t] = h;
    __syncthreads();
    for (int s = 128; s > 0; s >>= 1) {
        if (t < s) red[t] += red[t + s];
        __syncthreads();
    }
    if (t == 0) {
        out_perp[0] = expf(red[0]);
        out_loss[0] = COMMIT * num[0] / den[0];
    }
}

// ---------------------------------------------------------------------------
extern "C" void kernel_launch(void* const* d_in, const int* in_sizes, int n_in,
                              void* d_out, int out_size, void* d_ws, size_t ws_size,
                              hipStream_t stream) {
    const float* x   = (const float*)d_in[0];
    const float* emb = (const float*)d_in[1];
    const int B = in_sizes[0] / EMB_DIM;               // 65536

    // workspace layout (floats)
    float* ws_f   = (float*)d_ws;
    float* enorm  = ws_f;                              // [1024]
    float* counts = ws_f + 1024;                       // [1024]
    float* num    = ws_f + 2048;                       // [1]
    float* den    = ws_f + 2049;                       // [1]
    int*   idxbuf = (int*)(ws_f + 2304);               // [B]

    // output layout: quantized [B*256], loss [1], indices(float) [B], perplexity [1]
    float* out      = (float*)d_out;
    float* q_out    = out;
    float* out_loss = out + (size_t)B * EMB_DIM;
    float* out_idx  = out_loss + 1;
    float* out_perp = out_idx + B;

    vq_zero_kernel<<<1, 256, 0, stream>>>(counts);     // counts+num+den contiguous
    vq_enorm_kernel<<<N_EMB, 32, 0, stream>>>(emb, enorm);
    vq_argmin_wmma_kernel<<<B / 256, 256, 0, stream>>>(x, emb, enorm, idxbuf);
    vq_gather_loss_kernel<<<B / 8, 256, 0, stream>>>(x, emb, idxbuf, q_out, out_idx,
                                                     counts, num, den);
    vq_finalize_kernel<<<1, 256, 0, stream>>>(counts, num, den, out_loss, out_perp, B);
}
